// SkillExpertHead_52931176956050
// MI455X (gfx1250) — matrix operand order repaired and checked by reading, not compiled, not run on hardware
//
#include <hip/hip_runtime.h>
#include <hip/hip_bf16.h>
#include <math.h>

// ---------------------------------------------------------------------------
// Model dims
// ---------------------------------------------------------------------------
#define BB     64
#define TT     256
#define LTXT   64
#define DTXT   1024
#define DIMG   1024
#define DSTATE 32
#define DSKILL 64
#define DIN    2144   // 67*32
#define HH     1024
#define AA     32
#define VV     9
#define MTOT   (BB * TT)   // 16384

// ---------------------------------------------------------------------------
// CDNA5 async global->LDS path (ASYNCcnt) if the toolchain exposes it
// ---------------------------------------------------------------------------
#if defined(__gfx1250__) && __has_builtin(__builtin_amdgcn_global_load_async_to_lds_b128) && __has_builtin(__builtin_amdgcn_s_wait_asynccnt)
#define USE_ASYNC_LDS 1
#else
#define USE_ASYNC_LDS 0
#endif

// ---------------------------------------------------------------------------
// WMMA types / helpers
// ---------------------------------------------------------------------------
typedef __attribute__((ext_vector_type(16))) __bf16 v16bf;
typedef __attribute__((ext_vector_type(8)))  float  v8f;

union FragBF { uint4 u[2]; v16bf v; };

__device__ __forceinline__ unsigned short f2bf(float f) {
  unsigned int u = __float_as_uint(f);
  unsigned int r = u + 0x7FFFu + ((u >> 16) & 1u);   // round-to-nearest-even
  return (unsigned short)(r >> 16);
}
__device__ __forceinline__ float bf2f(unsigned short b) {
  return __uint_as_float(((unsigned int)b) << 16);
}
__device__ __forceinline__ float geluf(float x) {
  return 0.5f * x * (1.0f + erff(x * 0.70710678118654752f));
}
__device__ __forceinline__ float sigm(float x) { return 1.0f / (1.0f + __expf(-x)); }
__device__ __forceinline__ int imin(int a, int b) { return a < b ? a : b; }

#if USE_ASYNC_LDS
typedef int v4i_g __attribute__((vector_size(16)));
typedef __attribute__((address_space(1))) v4i_g* as1_v4i;
typedef __attribute__((address_space(3))) v4i_g* as3_v4i;

__device__ __forceinline__ void async_cp16(const unsigned short* g, unsigned short* l) {
  __builtin_amdgcn_global_load_async_to_lds_b128((as1_v4i)g, (as3_v4i)l, 0, 0);
}
#endif

// ---------------------------------------------------------------------------
// Generic bf16 WMMA GEMM:  out[M,N] = act(A[M,K] * Wt[N,K]^T + bias)
// Wt stored [N][K] (K contiguous).  Block tile 128x128, K-step 32, 8 waves.
// ---------------------------------------------------------------------------
#define BMT 128
#define BNT 128
#define BKT 32
#define LDT 40   // padded LDS row stride (ushorts): 80B, keeps uint4 alignment

template<int ACT, int WBF, int WF32, int BATCHED>
__global__ __launch_bounds__(256)
void gemm_wmma(const unsigned short* __restrict__ Ag,
               const unsigned short* __restrict__ Wt,
               const float* __restrict__ bias,
               unsigned short* __restrict__ outB,
               float* __restrict__ outF,
               int M, int N, int K, int ldOut,
               const int* __restrict__ skills,
               long wStride, long bStride, int Mb)
{
#if USE_ASYNC_LDS
  __shared__ alignas(16) unsigned short As[2][BMT * LDT];
  __shared__ alignas(16) unsigned short Bs[2][BNT * LDT];
#else
  __shared__ alignas(16) unsigned short As[1][BMT * LDT];
  __shared__ alignas(16) unsigned short Bs[1][BNT * LDT];
#endif

  const int tid  = threadIdx.x;
  const int lane = tid & 31;
  const int wid  = tid >> 5;
  const int wMi  = wid >> 2;          // 0..1  -> 64-row slab
  const int wNi  = wid & 3;           // 0..3  -> 32-col slab
  const int nBlk = blockIdx.x;
  const int mBlk = blockIdx.y;

  const int  bIdx    = BATCHED ? (int)blockIdx.z : 0;
  const int  Mloc    = BATCHED ? Mb : M;
  const long rowBase = BATCHED ? (long)bIdx * Mb : 0;

  const unsigned short* Wb    = Wt;
  const float*          biasb = bias;
  if (BATCHED) {
    int s = skills ? skills[bIdx] : 0;
    Wb += (long)s * wStride;
    if (biasb) biasb += (long)s * bStride;
  }

  v8f acc[4][2];
  const v8f z8 = {0.f,0.f,0.f,0.f,0.f,0.f,0.f,0.f};
#pragma unroll
  for (int i = 0; i < 4; ++i)
#pragma unroll
    for (int j = 0; j < 2; ++j) acc[i][j] = z8;

  const int ldRow = tid >> 1;         // 0..127
  const int ldSeg = (tid & 1) << 4;   // 0 or 16 (ushorts)
  const int aRow  = imin(mBlk * BMT + ldRow, Mloc - 1);   // clamped: garbage only
  const int bRow  = imin(nBlk * BNT + ldRow, N - 1);      // lands in masked out rows
  const unsigned short* aPtr = Ag + (rowBase + aRow) * (long)K + ldSeg;
  const unsigned short* bPtr = Wb + (long)bRow * K + ldSeg;

  const int g   = lane >> 4;   // K-half selector: lane half g holds k = 16*g .. 16*g+15 (contiguous order)
  const int l16 = lane & 15;

#if USE_ASYNC_LDS
  // -------- double-buffered async global->LDS pipeline (ASYNCcnt) ----------
  const int nk = K / BKT;
  {
    unsigned short* aD = &As[0][ldRow * LDT + ldSeg];
    unsigned short* bD = &Bs[0][ldRow * LDT + ldSeg];
    async_cp16(aPtr, aD);      async_cp16(aPtr + 8, aD + 8);
    async_cp16(bPtr, bD);      async_cp16(bPtr + 8, bD + 8);
  }
  for (int i = 0; i < nk; ++i) {
    __syncthreads();                       // all waves done reading buf[(i+1)&1]
    if (i + 1 < nk) {
      const int nb = (i + 1) & 1;
      const int k1 = (i + 1) * BKT;
      unsigned short* aD = &As[nb][ldRow * LDT + ldSeg];
      unsigned short* bD = &Bs[nb][ldRow * LDT + ldSeg];
      async_cp16(aPtr + k1, aD);     async_cp16(aPtr + k1 + 8, aD + 8);
      async_cp16(bPtr + k1, bD);     async_cp16(bPtr + k1 + 8, bD + 8);
      __builtin_amdgcn_s_wait_asynccnt(4);   // buf[i&1] done, next in flight
    } else {
      __builtin_amdgcn_s_wait_asynccnt(0);   // last tile: drain
    }
    __syncthreads();                         // buf[i&1] visible from all waves
    const int cb = i & 1;
    FragBF af[4], bfm[2];
#pragma unroll
    for (int mt = 0; mt < 4; ++mt) {
      const int r = wMi * 64 + mt * 16 + l16;
      af[mt].u[0] = *reinterpret_cast<const uint4*>(&As[cb][r * LDT + g * 16]);
      af[mt].u[1] = *reinterpret_cast<const uint4*>(&As[cb][r * LDT + g * 16 + 8]);
    }
#pragma unroll
    for (int nt = 0; nt < 2; ++nt) {
      const int n = wNi * 32 + nt * 16 + l16;
      const uint4* p = reinterpret_cast<const uint4*>(&Bs[cb][n * LDT + g * 16]);
      bfm[nt].u[0] = p[0]; bfm[nt].u[1] = p[1];
    }
#pragma unroll
    for (int mt = 0; mt < 4; ++mt)
#pragma unroll
      for (int nt = 0; nt < 2; ++nt)
        acc[mt][nt] = __builtin_amdgcn_wmma_f32_16x16x32_bf16(
            false, af[mt].v, false, bfm[nt].v, (short)0, acc[mt][nt], false, false);
  }
#else
  // -------- fallback: register-staged, single-buffered ---------------------
  for (int k0 = 0; k0 < K; k0 += BKT) {
    const uint4* pa = reinterpret_cast<const uint4*>(aPtr + k0);
    const uint4* pb = reinterpret_cast<const uint4*>(bPtr + k0);
    uint4 a0 = pa[0], a1 = pa[1];
    uint4 b0 = pb[0], b1 = pb[1];
    if (k0 + BKT < K) {
      __builtin_prefetch(aPtr + k0 + BKT, 0, 3);
      __builtin_prefetch(bPtr + k0 + BKT, 0, 3);
    }
    __syncthreads();
    *reinterpret_cast<uint4*>(&As[0][ldRow * LDT + ldSeg])     = a0;
    *reinterpret_cast<uint4*>(&As[0][ldRow * LDT + ldSeg + 8]) = a1;
    *reinterpret_cast<uint4*>(&Bs[0][ldRow * LDT + ldSeg])     = b0;
    *reinterpret_cast<uint4*>(&Bs[0][ldRow * LDT + ldSeg + 8]) = b1;
    __syncthreads();

    FragBF af[4], bfm[2];
#pragma unroll
    for (int mt = 0; mt < 4; ++mt) {
      const int r = wMi * 64 + mt * 16 + l16;
      af[mt].u[0] = *reinterpret_cast<const uint4*>(&As[0][r * LDT + g * 16]);
      af[mt].u[1] = *reinterpret_cast<const uint4*>(&As[0][r * LDT + g * 16 + 8]);
    }
#pragma unroll
    for (int nt = 0; nt < 2; ++nt) {
      const int n = wNi * 32 + nt * 16 + l16;
      const uint4* p = reinterpret_cast<const uint4*>(&Bs[0][n * LDT + g * 16]);
      bfm[nt].u[0] = p[0]; bfm[nt].u[1] = p[1];
    }
#pragma unroll
    for (int mt = 0; mt < 4; ++mt)
#pragma unroll
      for (int nt = 0; nt < 2; ++nt)
        acc[mt][nt] = __builtin_amdgcn_wmma_f32_16x16x32_bf16(
            false, af[mt].v, false, bfm[nt].v, (short)0, acc[mt][nt], false, false);
  }
#endif

  // Epilogue: C/D layout — lane<16: M=v, N=lane ; lane>=16: M=v+8, N=lane-16
#pragma unroll
  for (int nt = 0; nt < 2; ++nt) {
    const int gcol = nBlk * BNT + wNi * 32 + nt * 16 + l16;
    if (gcol >= N) continue;
    const float bv = biasb ? biasb[gcol] : 0.0f;
#pragma unroll
    for (int mt = 0; mt < 4; ++mt) {
#pragma unroll
      for (int v = 0; v < 8; ++v) {
        const int grow = mBlk * BMT + wMi * 64 + mt * 16 + g * 8 + v;
        if (grow < Mloc) {
          float x = acc[mt][nt][v] + bv;
          if (ACT == 1) x = geluf(x);
          const long o = (rowBase + grow) * (long)ldOut + gcol;
          if (WF32) outF[o] = x;
          if (WBF)  outB[o] = f2bf(x);
        }
      }
    }
  }
}

// ---------------------------------------------------------------------------
// Persistent GRU kernel: 8 blocks (one per 128-column slice of H), loops over
// all T=256 steps.  Per step each block computes its slice of all 3 gates
// (three 64x128x1024 bf16 WMMA GEMMs), fuses the torch-GRU gate math in the
// epilogue, and synchronizes via a monotonic atomic grid barrier.
// h (f32) is per-column owned; hbf is ping-ponged [2][BB*HH] for the A-matrix.
// ---------------------------------------------------------------------------
#define GLDT 40

__global__ __launch_bounds__(256)
void gru_persistent_kernel(const unsigned short* __restrict__ whhB,  // [3H][H] bf16
                           const float* __restrict__ gi,             // [MTOT][3H] (incl. bih)
                           const float* __restrict__ bhh,            // [3H]
                           float* __restrict__ h,                    // [BB*HH] f32 (zeroed)
                           unsigned short* __restrict__ hbf,         // [2][BB*HH] (buf0 zeroed)
                           unsigned short* __restrict__ xg,          // [MTOT][HH] bf16 out
                           unsigned int* __restrict__ bar)           // zeroed counter
{
#if USE_ASYNC_LDS
  __shared__ alignas(16) unsigned short Ah[2][64 * GLDT];
  __shared__ alignas(16) unsigned short Bw[2][384 * GLDT];
#else
  __shared__ alignas(16) unsigned short Ah[1][64 * GLDT];
  __shared__ alignas(16) unsigned short Bw[1][384 * GLDT];
#endif
  const int tid  = threadIdx.x;
  const int lane = tid & 31;
  const int wid  = tid >> 5;          // 0..7 -> 16-column slab within the 128
  const int g    = lane >> 4;
  const int l16  = lane & 15;
  const int c0   = blockIdx.x * 128;  // block's H-column base
  const int nl   = wid * 16;

  // ---- loader mapping: 16-byte chunks, uniform 7 async issues per thread ----
  // A tile 64x32 ushorts = 256 chunks -> 1 per thread
  const int aRow = tid >> 2;               // 0..63
  const int aSeg = (tid & 3) << 3;         // 0,8,16,24 (ushorts)
  // B tile 384x32 ushorts = 1536 chunks -> 6 per thread
  const unsigned short* bPtrQ[6];
  int bRowQ[6], bSegQ[6];
#pragma unroll
  for (int q = 0; q < 6; ++q) {
    const int chunk = tid + 256 * q;       // 0..1535
    const int brow  = chunk >> 2;          // 0..383 (= gate*128 + colLocal)
    const int bseg  = (chunk & 3) << 3;
    const int gate  = brow >> 7;
    const int colL  = brow & 127;
    bPtrQ[q] = whhB + ((long)gate * HH + c0 + colL) * HH + bseg;
    bRowQ[q] = brow; bSegQ[q] = bseg;
  }

  v8f acc[3][4];
  const v8f z8 = {0.f,0.f,0.f,0.f,0.f,0.f,0.f,0.f};

  const float bhr = bhh[c0 + nl + l16];
  const float bhz = bhh[HH + c0 + nl + l16];
  const float bhn = bhh[2 * HH + c0 + nl + l16];
  const int   c   = c0 + nl + l16;

  for (int t = 0; t < TT; ++t) {
#pragma unroll
    for (int x = 0; x < 3; ++x)
#pragma unroll
      for (int mt = 0; mt < 4; ++mt) acc[x][mt] = z8;

    const unsigned short* aG = hbf + (size_t)(t & 1) * (BB * HH) + (long)aRow * HH + aSeg;

#if USE_ASYNC_LDS
    {
      async_cp16(aG, &Ah[0][aRow * GLDT + aSeg]);
#pragma unroll
      for (int q = 0; q < 6; ++q)
        async_cp16(bPtrQ[q], &Bw[0][bRowQ[q] * GLDT + bSegQ[q]]);
    }
    for (int i = 0; i < HH / BKT; ++i) {
      __syncthreads();
      if (i + 1 < HH / BKT) {
        const int nb = (i + 1) & 1;
        const int k1 = (i + 1) * BKT;
        async_cp16(aG + k1, &Ah[nb][aRow * GLDT + aSeg]);
#pragma unroll
        for (int q = 0; q < 6; ++q)
          async_cp16(bPtrQ[q] + k1, &Bw[nb][bRowQ[q] * GLDT + bSegQ[q]]);
        __builtin_amdgcn_s_wait_asynccnt(7);
      } else {
        __builtin_amdgcn_s_wait_asynccnt(0);
      }
      __syncthreads();
      const int cb = i & 1;
      FragBF af[4];
#pragma unroll
      for (int mt = 0; mt < 4; ++mt) {
        const int r = mt * 16 + l16;
        af[mt].u[0] = *reinterpret_cast<const uint4*>(&Ah[cb][r * GLDT + g * 16]);
        af[mt].u[1] = *reinterpret_cast<const uint4*>(&Ah[cb][r * GLDT + g * 16 + 8]);
      }
#pragma unroll
      for (int x = 0; x < 3; ++x) {
        FragBF bfm;
        const uint4* p = reinterpret_cast<const uint4*>(&Bw[cb][(x * 128 + nl + l16) * GLDT + g * 16]);
        bfm.u[0] = p[0]; bfm.u[1] = p[1];
#pragma unroll
        for (int mt = 0; mt < 4; ++mt)
          acc[x][mt] = __builtin_amdgcn_wmma_f32_16x16x32_bf16(
              false, af[mt].v, false, bfm.v, (short)0, acc[x][mt], false, false);
      }
    }
#else
    for (int i = 0; i < HH / BKT; ++i) {
      const int k0 = i * BKT;
      uint4 av = *reinterpret_cast<const uint4*>(aG + k0);
      uint4 bv[6];
#pragma unroll
      for (int q = 0; q < 6; ++q) bv[q] = *reinterpret_cast<const uint4*>(bPtrQ[q] + k0);
      __syncthreads();
      *reinterpret_cast<uint4*>(&Ah[0][aRow * GLDT + aSeg]) = av;
#pragma unroll
      for (int q = 0; q < 6; ++q)
        *reinterpret_cast<uint4*>(&Bw[0][bRowQ[q] * GLDT + bSegQ[q]]) = bv[q];
      __syncthreads();
      FragBF af[4];
#pragma unroll
      for (int mt = 0; mt < 4; ++mt) {
        const int r = mt * 16 + l16;
        af[mt].u[0] = *reinterpret_cast<const uint4*>(&Ah[0][r * GLDT + g * 16]);
        af[mt].u[1] = *reinterpret_cast<const uint4*>(&Ah[0][r * GLDT + g * 16 + 8]);
      }
#pragma unroll
      for (int x = 0; x < 3; ++x) {
        FragBF bfm;
        const uint4* p = reinterpret_cast<const uint4*>(&Bw[0][(x * 128 + nl + l16) * GLDT + g * 16]);
        bfm.u[0] = p[0]; bfm.u[1] = p[1];
#pragma unroll
        for (int mt = 0; mt < 4; ++mt)
          acc[x][mt] = __builtin_amdgcn_wmma_f32_16x16x32_bf16(
              false, af[mt].v, false, bfm.v, (short)0, acc[x][mt], false, false);
      }
    }
#endif

    // ---- fused GRU gate epilogue: rows b = mt*16 + g*8 + v, column c --------
    unsigned short* hbfNext = hbf + (size_t)((t + 1) & 1) * (BB * HH);
#pragma unroll
    for (int mt = 0; mt < 4; ++mt) {
#pragma unroll
      for (int v = 0; v < 8; ++v) {
        const int b = mt * 16 + g * 8 + v;
        const long gio = ((long)b * TT + t) * (3 * HH);
        const float r = sigm(gi[gio + c]          + acc[0][mt][v] + bhr);
        const float z = sigm(gi[gio + HH + c]     + acc[1][mt][v] + bhz);
        const float n = tanhf(gi[gio + 2 * HH + c] + r * (acc[2][mt][v] + bhn));
        const int hidx = b * HH + c;
        const float hp = h[hidx];
        const float hv = (1.0f - z) * n + z * hp;
        h[hidx] = hv;
        const unsigned short bb = f2bf(hv);
        hbfNext[hidx] = bb;
        xg[((long)b * TT + t) * HH + c] = bb;
      }
    }

    // ---- grid barrier (monotonic counter, 8 co-resident blocks) -------------
    __threadfence();
    __syncthreads();
    if (tid == 0) {
      atomicAdd(bar, 1u);
      const unsigned target = (unsigned)(t + 1) * gridDim.x;
      while (__hip_atomic_load(bar, __ATOMIC_ACQUIRE, __HIP_MEMORY_SCOPE_AGENT) < target) {
        __builtin_amdgcn_s_sleep(2);
      }
    }
    __syncthreads();
  }
}

// ---------------------------------------------------------------------------
// Elementwise / prep kernels
// ---------------------------------------------------------------------------
__global__ void txt_mean_kernel(const float* __restrict__ txt, float* __restrict__ out) {
  int idx = blockIdx.x * blockDim.x + threadIdx.x;        // BB*DTXT
  if (idx >= BB * DTXT) return;
  int b = idx >> 10, d = idx & 1023;
  const float* p = txt + (long)b * LTXT * DTXT + d;
  float s = 0.f;
#pragma unroll 4
  for (int l = 0; l < LTXT; ++l) s += p[(long)l * DTXT];
  out[idx] = s * (1.0f / LTXT);
}

__global__ void build_xin_kernel(const float* __restrict__ tmean,
                                 const float* __restrict__ img,
                                 const float* __restrict__ state,
                                 const int*   __restrict__ skills,
                                 const float* __restrict__ emb,
                                 unsigned short* __restrict__ xin) {
  long idx = (long)blockIdx.x * blockDim.x + threadIdx.x;
  if (idx >= (long)MTOT * DIN) return;
  int c = (int)(idx % DIN);
  long m = idx / DIN;
  int b = (int)(m >> 8), t = (int)(m & 255);
  float v;
  if (c < DTXT)                 v = tmean[(long)b * DTXT + c];
  else if (c < DTXT + DIMG)     v = img[((long)b * TT + t) * DIMG + (c - DTXT)];
  else if (c < DTXT + DIMG + DSTATE) v = state[((long)b * TT + t) * DSTATE + (c - DTXT - DIMG)];
  else                          v = emb[(long)skills[b] * DSKILL + (c - DTXT - DIMG - DSTATE)];
  xin[idx] = f2bf(v);
}

__global__ void cast_bf16_kernel(const float* __restrict__ in, unsigned short* __restrict__ out, long n) {
  long i = (long)blockIdx.x * blockDim.x + threadIdx.x;
  if (i < n) out[i] = f2bf(in[i]);
}

// in [Bt][R][C] f32  ->  out [Bt][C][R] bf16
__global__ void transpose_bf16_kernel(const float* __restrict__ in, unsigned short* __restrict__ out,
                                      int Bt, int R, int C) {
  long i = (long)blockIdx.x * blockDim.x + threadIdx.x;
  long total = (long)Bt * R * C;
  if (i >= total) return;
  long rc = i % ((long)R * C);
  int b = (int)(i / ((long)R * C));
  int r = (int)(rc / C), c = (int)(rc % C);
  out[((long)b * C + c) * R + r] = f2bf(in[i]);
}

__global__ void zero_init_kernel(float* __restrict__ h, unsigned short* __restrict__ hbf,
                                 unsigned int* __restrict__ bar) {
  int i = blockIdx.x * blockDim.x + threadIdx.x;
  if (i < BB * HH) h[i] = 0.0f;
  if (i < 2 * BB * HH) hbf[i] = 0;
  if (i == 0) bar[0] = 0u;
}

// p[row] = sigmoid(dot(Dh[row,0:512], w2) + b2) ; one wave per row
__global__ void done2_kernel(const unsigned short* __restrict__ dh,
                             const float* __restrict__ w2, const float* __restrict__ b2,
                             float* __restrict__ p) {
  int row  = blockIdx.x * 8 + (threadIdx.x >> 5);
  int lane = threadIdx.x & 31;
  if (row >= MTOT) return;
  const unsigned short* x = dh + (long)row * (HH / 2);
  float s = 0.f;
  for (int i = lane; i < HH / 2; i += 32) s += bf2f(x[i]) * w2[i];
  for (int off = 16; off; off >>= 1) s += __shfl_xor(s, off, 32);
  if (lane == 0) p[row] = sigm(s + b2[0]);
}

// ---------------------------------------------------------------------------
// Host launcher
// ---------------------------------------------------------------------------
static inline dim3 g1(long n, int blk) { return dim3((unsigned)((n + blk - 1) / blk)); }

extern "C" void kernel_launch(void* const* d_in, const int* in_sizes, int n_in,
                              void* d_out, int out_size, void* d_ws, size_t ws_size,
                              hipStream_t stream) {
  const float* txt     = (const float*)d_in[0];
  const float* img     = (const float*)d_in[1];
  const float* state   = (const float*)d_in[2];
  const int*   skills  = (const int*)  d_in[3];
  const float* emb     = (const float*)d_in[4];
  const float* fuse_W  = (const float*)d_in[5];
  const float* fuse_b  = (const float*)d_in[6];
  const float* gru_Wih = (const float*)d_in[7];
  const float* gru_Whh = (const float*)d_in[8];
  const float* gru_bih = (const float*)d_in[9];
  const float* gru_bhh = (const float*)d_in[10];
  const float* ad_W1   = (const float*)d_in[11];
  const float* ad_b1   = (const float*)d_in[12];
  const float* ad_W2   = (const float*)d_in[13];
  const float* ad_b2   = (const float*)d_in[14];
  const float* head_W1 = (const float*)d_in[15];
  const float* head_b1 = (const float*)d_in[16];
  const float* head_W2 = (const float*)d_in[17];
  const float* head_b2 = (const float*)d_in[18];
  const float* done_W1 = (const float*)d_in[19];
  const float* done_b1 = (const float*)d_in[20];
  const float* done_W2 = (const float*)d_in[21];
  const float* done_b2 = (const float*)d_in[22];

  // bump allocator over workspace
  char* ws = (char*)d_ws;
  size_t off = 0;
  auto alloc = [&](size_t bytes) -> void* {
    off = (off + 255) & ~(size_t)255;
    void* p = ws + off;
    off += bytes;
    return p;
  };
  float*          tmean   = (float*)alloc((size_t)BB * DTXT * 4);
  unsigned short* xin     = (unsigned short*)alloc((size_t)MTOT * DIN * 2);
  unsigned short* fuseWt  = (unsigned short*)alloc((size_t)HH * DIN * 2);
  unsigned short* xf      = (unsigned short*)alloc((size_t)MTOT * HH * 2);
  unsigned short* wihB    = (unsigned short*)alloc((size_t)3 * HH * HH * 2);
  unsigned short* whhB    = (unsigned short*)alloc((size_t)3 * HH * HH * 2);
  float*          gi      = (float*)alloc((size_t)MTOT * 3 * HH * 4);
  float*          hbuf    = (float*)alloc((size_t)BB * HH * 4);
  unsigned short* hbf     = (unsigned short*)alloc((size_t)2 * BB * HH * 2);
  unsigned int*   bar     = (unsigned int*)alloc(256);
  unsigned short* xg      = (unsigned short*)alloc((size_t)MTOT * HH * 2);
  unsigned short* adW1t   = (unsigned short*)alloc((size_t)VV * HH * HH * 2);
  unsigned short* adW2t   = (unsigned short*)alloc((size_t)VV * HH * HH * 2);
  unsigned short* ybuf    = (unsigned short*)alloc((size_t)MTOT * HH * 2);
  unsigned short* xa      = (unsigned short*)alloc((size_t)MTOT * HH * 2);
  unsigned short* headW1t = (unsigned short*)alloc((size_t)HH * HH * 2);
  unsigned short* headW2t = (unsigned short*)alloc((size_t)AA * HH * 2);
  unsigned short* hh1     = (unsigned short*)alloc((size_t)MTOT * HH * 2);
  unsigned short* doneW1t = (unsigned short*)alloc((size_t)(HH / 2) * HH * 2);
  unsigned short* dh      = (unsigned short*)alloc((size_t)MTOT * (HH / 2) * 2);

  float* outA = (float*)d_out;               // [B,T,A]
  float* outP = outA + (size_t)MTOT * AA;    // [B,T]

  const int BLK = 256;

  // ---- prep & weight conversion ----
  txt_mean_kernel<<<g1((long)BB * DTXT, BLK), BLK, 0, stream>>>(txt, tmean);
  transpose_bf16_kernel<<<g1((long)DIN * HH, BLK), BLK, 0, stream>>>(fuse_W, fuseWt, 1, DIN, HH);
  cast_bf16_kernel<<<g1((long)3 * HH * HH, BLK), BLK, 0, stream>>>(gru_Wih, wihB, (long)3 * HH * HH);
  cast_bf16_kernel<<<g1((long)3 * HH * HH, BLK), BLK, 0, stream>>>(gru_Whh, whhB, (long)3 * HH * HH);
  transpose_bf16_kernel<<<g1((long)VV * HH * HH, BLK), BLK, 0, stream>>>(ad_W1, adW1t, VV, HH, HH);
  transpose_bf16_kernel<<<g1((long)VV * HH * HH, BLK), BLK, 0, stream>>>(ad_W2, adW2t, VV, HH, HH);
  transpose_bf16_kernel<<<g1((long)HH * HH, BLK), BLK, 0, stream>>>(head_W1, headW1t, 1, HH, HH);
  transpose_bf16_kernel<<<g1((long)HH * AA, BLK), BLK, 0, stream>>>(head_W2, headW2t, 1, HH, AA);
  transpose_bf16_kernel<<<g1((long)HH * (HH / 2), BLK), BLK, 0, stream>>>(done_W1, doneW1t, 1, HH, HH / 2);
  build_xin_kernel<<<g1((long)MTOT * DIN, BLK), BLK, 0, stream>>>(tmean, img, state, skills, emb, xin);
  zero_init_kernel<<<g1((long)2 * BB * HH, BLK), BLK, 0, stream>>>(hbuf, hbf, bar);

  // ---- fuse: Xf = gelu(Xin @ fuse_W + b)   [16384 x 1024], K=2144 ----
  gemm_wmma<1,1,0,0><<<dim3(HH / BNT, MTOT / BMT), BLK, 0, stream>>>(
      xin, fuseWt, fuse_b, xf, nullptr, MTOT, HH, DIN, HH, nullptr, 0, 0, 0);

  // ---- gi = Xf @ Wih^T + bih   [16384 x 3072], K=1024 ----
  gemm_wmma<0,0,1,0><<<dim3(3 * HH / BNT, MTOT / BMT), BLK, 0, stream>>>(
      xf, wihB, gru_bih, nullptr, gi, MTOT, 3 * HH, HH, 3 * HH, nullptr, 0, 0, 0);

  // ---- GRU recurrence: single persistent kernel, 8 co-resident blocks ----
  gru_persistent_kernel<<<dim3(8), BLK, 0, stream>>>(whhB, gi, gru_bhh, hbuf, hbf, xg, bar);

  // ---- skill adapter (batched per-sample weights) ----
  gemm_wmma<1,1,0,1><<<dim3(HH / BNT, TT / BMT, BB), BLK, 0, stream>>>(
      xg, adW1t, ad_b1, ybuf, nullptr, MTOT, HH, HH, HH, skills, (long)HH * HH, (long)HH, TT);
  gemm_wmma<0,1,0,1><<<dim3(HH / BNT, TT / BMT, BB), BLK, 0, stream>>>(
      ybuf, adW2t, ad_b2, xa, nullptr, MTOT, HH, HH, HH, skills, (long)HH * HH, (long)HH, TT);

  // ---- action head ----
  gemm_wmma<1,1,0,0><<<dim3(HH / BNT, MTOT / BMT), BLK, 0, stream>>>(
      xa, headW1t, head_b1, hh1, nullptr, MTOT, HH, HH, HH, nullptr, 0, 0, 0);
  gemm_wmma<0,0,1,0><<<dim3(1, MTOT / BMT), BLK, 0, stream>>>(
      hh1, headW2t, head_b2, nullptr, outA, MTOT, AA, HH, AA, nullptr, 0, 0, 0);

  // ---- done head ----
  gemm_wmma<1,1,0,0><<<dim3((HH / 2) / BNT, MTOT / BMT), BLK, 0, stream>>>(
      xa, doneW1t, done_b1, dh, nullptr, MTOT, HH / 2, HH, HH / 2, nullptr, 0, 0, 0);
  done2_kernel<<<g1((long)MTOT, 8), 256, 0, stream>>>(dh, done_W2, done_b2, outP);

  (void)in_sizes; (void)n_in; (void)out_size; (void)ws_size;
}
